// TriangleMultiplication_36902359007375
// MI455X (gfx1250) — compile-verified, matrix operand order checked by reading, not run on hardware
//
#include <hip/hip_runtime.h>
#include <hip/hip_bf16.h>

typedef __attribute__((ext_vector_type(16))) _Float16 v16h;
typedef __attribute__((ext_vector_type(8)))  _Float16 v8h;
typedef __attribute__((ext_vector_type(8)))  float    v8f;

#define NDIM 512
#define CDIM 128
#define NN   (512 * 512)

__device__ __forceinline__ int kidx_of(int lane, int e) {
    // 16-bit A/B fragment K index: lanes 0-15 -> K {0..7, 16..23}, lanes 16-31 -> K {8..15, 24..31}
    return ((lane >> 4) << 3) + (e < 8 ? e : e + 8);
}

__device__ __forceinline__ float sigmoidf_(float x) {
    // v_exp_f32 + v_rcp_f32 (1-ulp approx is far inside fp16-pipeline tolerance)
    return __builtin_amdgcn_rcpf(1.0f + __expf(-x));
}

// Load a 16x32 f16 A (or 32x16 B) fragment from a row-major (row, k) half matrix.
__device__ __forceinline__ v16h load_frag_rows(const _Float16* __restrict__ base,
                                               int row_stride, int rowbase,
                                               int lane, int k0) {
    int r = rowbase + (lane & 15);
    const _Float16* rp = base + (size_t)r * row_stride + k0 + ((lane >> 4) << 3);
    v8h lo = *(const v8h*)(rp);       // K = k0+8h .. +7
    v8h hi = *(const v8h*)(rp + 16);  // K = k0+8h+16 .. +23
    return __builtin_shufflevector(lo, hi, 0,1,2,3,4,5,6,7,8,9,10,11,12,13,14,15);
}

// Load a pre-packed weight B-fragment: one contiguous 32B chunk per lane.
__device__ __forceinline__ v16h load_frag_packed(const _Float16* __restrict__ pk,
                                                 int fragIdx, int lane) {
    return *(const v16h*)(pk + ((size_t)fragIdx * 32 + lane) * 16);
}

// ---------------------------------------------------------------------------
// Kernel 0: pack a (K=128, D) f32 weight into WMMA B-fragment f16 layout.
// fragIdx = dtile*4 + kstep ; element t -> pk[t] = w[k*D + d]
// ---------------------------------------------------------------------------
__global__ void pack_w(const float* __restrict__ w, _Float16* __restrict__ pk,
                       int D, int total) {
    int t = blockIdx.x * blockDim.x + threadIdx.x;
    if (t >= total) return;
    int e     = t & 15;
    int lane  = (t >> 4) & 31;
    int f     = t >> 9;
    int kstep = f & 3;
    int dtile = f >> 2;
    int k = kstep * 32 + kidx_of(lane, e);
    int d = dtile * 16 + (lane & 15);
    pk[t] = (_Float16)w[(size_t)k * D + d];
}

// ---------------------------------------------------------------------------
// Kernel 1: per wave = 16 positions. LayerNorm -> A fragments ->
//   proj = (x@w_proj) * sigmoid(x@w_gate) * mask  -> a/b f16 (C,N,N)
//   gate_out = sigmoid(x@w_gate_out + b_gate_out) -> f32 (C, N*N)
// ---------------------------------------------------------------------------
__global__ __launch_bounds__(32) void k1_ln_proj(
    const float* __restrict__ act, const float* __restrict__ mask,
    const float* __restrict__ ln_scale, const float* __restrict__ ln_offset,
    const float* __restrict__ b_gate_out,
    const _Float16* __restrict__ pk_proj, const _Float16* __restrict__ pk_gate,
    const _Float16* __restrict__ pk_go,
    _Float16* __restrict__ abuf, _Float16* __restrict__ bbuf,
    float* __restrict__ gatebuf)
{
    __shared__ float xs[16][CDIM];
    __shared__ float lnS[CDIM], lnO[CDIM];
    __shared__ float maskS[16], meanS[16], rstdS[16];

    const int tid   = threadIdx.x;   // block = 1 wave
    const int lane  = tid & 31;
    const int pbase = blockIdx.x * 16;

    for (int i = tid; i < CDIM; i += 32) { lnS[i] = ln_scale[i]; lnO[i] = ln_offset[i]; }
    if (tid < 16) maskS[tid] = mask[pbase + tid];

    {   // load 16x128 f32 tile of act
        int row = tid >> 1, hsel = tid & 1;
        const float4* src = (const float4*)(act + (size_t)(pbase + row) * CDIM + hsel * 64);
        float4* dst = (float4*)(&xs[row][hsel * 64]);
        #pragma unroll
        for (int t = 0; t < 16; ++t) dst[t] = src[t];
    }
    __syncthreads();

    if (tid < 16) {
        float s = 0.f, s2 = 0.f;
        #pragma unroll 4
        for (int c = 0; c < CDIM; ++c) { float v = xs[tid][c]; s += v; s2 += v * v; }
        float m  = s  * (1.0f / CDIM);
        float vv = s2 * (1.0f / CDIM) - m * m;
        meanS[tid] = m;
        rstdS[tid] = rsqrtf(vv + 1e-5f);
    }
    __syncthreads();

    const int m    = lane & 15;
    const int half = lane >> 4;
    const float mn = meanS[m], rs = rstdS[m];

    v16h af[4];
    #pragma unroll
    for (int kk = 0; kk < 4; ++kk) {
        v16h a;
        #pragma unroll
        for (int e = 0; e < 16; ++e) {
            int c = kk * 32 + kidx_of(lane, e);
            float v = (xs[m][c] - mn) * rs * lnS[c] + lnO[c];
            a[e] = (_Float16)v;
        }
        af[kk] = a;
    }

    float mk[8];
    #pragma unroll
    for (int r = 0; r < 8; ++r) mk[r] = maskS[half * 8 + r];

    // projection + gate -> a / b buffers
    for (int dt = 0; dt < 16; ++dt) {
        v8f ap = {}, ag = {};
        #pragma unroll
        for (int kk = 0; kk < 4; ++kk) {
            v16h bp = load_frag_packed(pk_proj, dt * 4 + kk, lane);
            v16h bg = load_frag_packed(pk_gate, dt * 4 + kk, lane);
            ap = __builtin_amdgcn_wmma_f32_16x16x32_f16(false, af[kk], false, bp, (short)0, ap, false, false);
            ag = __builtin_amdgcn_wmma_f32_16x16x32_f16(false, af[kk], false, bg, (short)0, ag, false, false);
        }
        int d = dt * 16 + (lane & 15);
        _Float16* dstbase = ((d & 1) ? bbuf : abuf) + (size_t)(d >> 1) * NN + pbase + half * 8;
        v8h outv;
        #pragma unroll
        for (int r = 0; r < 8; ++r) outv[r] = (_Float16)(ap[r] * sigmoidf_(ag[r]) * mk[r]);
        *(v8h*)dstbase = outv;
    }

    // gate_out = sigmoid(x @ w_gate_out + b_gate_out), stored d-major (C, N*N)
    for (int dt = 0; dt < 8; ++dt) {
        v8f acc = {};
        #pragma unroll
        for (int kk = 0; kk < 4; ++kk) {
            v16h bg = load_frag_packed(pk_go, dt * 4 + kk, lane);
            acc = __builtin_amdgcn_wmma_f32_16x16x32_f16(false, af[kk], false, bg, (short)0, acc, false, false);
        }
        int d = dt * 16 + (lane & 15);
        float bias = b_gate_out[d];
        v8f gv;
        #pragma unroll
        for (int r = 0; r < 8; ++r) gv[r] = sigmoidf_(acc[r] + bias);
        *(v8f*)(gatebuf + (size_t)d * NN + pbase + half * 8) = gv;
    }
}

// ---------------------------------------------------------------------------
// Kernel 2: act2[c] = A_c (512x512) @ B_c^T, f16 in / f32 out.
// Block = 4 waves -> 128x128 tile; wave -> 64x64 (4x4 WMMA tiles), K = 512.
// 8 fragment loads feed 16 WMMAs per K-step (1 vector load per WMMA).
// ---------------------------------------------------------------------------
__global__ __launch_bounds__(128) void k2_pair_gemm(
    const _Float16* __restrict__ abuf, const _Float16* __restrict__ bbuf,
    float* __restrict__ act2)
{
    const int lane = threadIdx.x & 31;
    const int w    = threadIdx.x >> 5;
    const int c    = blockIdx.z;
    const int ibase = blockIdx.x * 128 + (w >> 1) * 64;
    const int jbase = blockIdx.y * 128 + (w & 1) * 64;

    const _Float16* A = abuf + (size_t)c * NN;
    const _Float16* B = bbuf + (size_t)c * NN;

    v8f acc[4][4] = {};
    #pragma unroll 1
    for (int kk = 0; kk < 16; ++kk) {
        int k0 = kk * 32;
        v16h a[4], b[4];
        #pragma unroll
        for (int t = 0; t < 4; ++t) {
            a[t] = load_frag_rows(A, NDIM, ibase + t * 16, lane, k0);
            b[t] = load_frag_rows(B, NDIM, jbase + t * 16, lane, k0);  // B[k][n] = b[jbase+n][k]
        }
        #pragma unroll
        for (int ti = 0; ti < 4; ++ti)
            #pragma unroll
            for (int tj = 0; tj < 4; ++tj)
                acc[ti][tj] = __builtin_amdgcn_wmma_f32_16x16x32_f16(
                    false, a[ti], false, b[tj], (short)0, acc[ti][tj], false, false);
    }

    const int n = lane & 15, half = lane >> 4;
    float* dst = act2 + (size_t)c * NN;
    #pragma unroll
    for (int ti = 0; ti < 4; ++ti)
        #pragma unroll
        for (int tj = 0; tj < 4; ++tj)
            #pragma unroll
            for (int r = 0; r < 8; ++r) {
                int i = ibase + ti * 16 + r + 8 * half;
                int j = jbase + tj * 16 + n;
                dst[(size_t)i * NDIM + j] = acc[ti][tj][r];
            }
}

// ---------------------------------------------------------------------------
// Kernel 3: channel-norm over c, then out = (y @ w_out + b_out) * gate_out.
// ---------------------------------------------------------------------------
__global__ __launch_bounds__(32) void k3_cnorm_out(
    const float* __restrict__ act2, const float* __restrict__ gatebuf,
    const float* __restrict__ cn_scale, const float* __restrict__ cn_offset,
    const float* __restrict__ b_out, const _Float16* __restrict__ pk_out,
    float* __restrict__ out)
{
    __shared__ float ys[16][CDIM];
    __shared__ float cnS[CDIM], cnO[CDIM];
    __shared__ float meanS[16], rstdS[16];

    const int tid   = threadIdx.x;
    const int lane  = tid & 31;
    const int pbase = blockIdx.x * 16;

    for (int i = tid; i < CDIM; i += 32) { cnS[i] = cn_scale[i]; cnO[i] = cn_offset[i]; }

    {   // gather 128 channels x 16 positions, transpose into pos-major LDS
        int pcol = tid & 15;
        int coff = tid >> 4;  // 0 or 1
        for (int c2 = 0; c2 < 64; ++c2) {
            int cc = c2 * 2 + coff;
            ys[pcol][cc] = act2[(size_t)cc * NN + pbase + pcol];
        }
    }
    __syncthreads();

    if (tid < 16) {
        float s = 0.f, s2 = 0.f;
        #pragma unroll 4
        for (int c = 0; c < CDIM; ++c) { float v = ys[tid][c]; s += v; s2 += v * v; }
        float m  = s  * (1.0f / CDIM);
        float vv = s2 * (1.0f / CDIM) - m * m;
        meanS[tid] = m;
        rstdS[tid] = rsqrtf(vv + 1e-5f);
    }
    __syncthreads();

    const int m    = lane & 15;
    const int half = lane >> 4;
    const float mn = meanS[m], rs = rstdS[m];

    v16h af[4];
    #pragma unroll
    for (int kk = 0; kk < 4; ++kk) {
        v16h a;
        #pragma unroll
        for (int e = 0; e < 16; ++e) {
            int c = kk * 32 + kidx_of(lane, e);
            float v = (ys[m][c] - mn) * rs * cnS[c] + cnO[c];
            a[e] = (_Float16)v;
        }
        af[kk] = a;
    }

    for (int dt = 0; dt < 8; ++dt) {
        v8f acc = {};
        #pragma unroll
        for (int kk = 0; kk < 4; ++kk) {
            v16h bw = load_frag_packed(pk_out, dt * 4 + kk, lane);
            acc = __builtin_amdgcn_wmma_f32_16x16x32_f16(false, af[kk], false, bw, (short)0, acc, false, false);
        }
        int d = dt * 16 + (lane & 15);
        float bias = b_out[d];
        v8f gv = *(const v8f*)(gatebuf + (size_t)d * NN + pbase + half * 8);
        #pragma unroll
        for (int r = 0; r < 8; ++r) {
            int p = pbase + half * 8 + r;
            out[(size_t)p * CDIM + d] = (acc[r] + bias) * gv[r];
        }
    }
}

// ---------------------------------------------------------------------------
extern "C" void kernel_launch(void* const* d_in, const int* in_sizes, int n_in,
                              void* d_out, int out_size, void* d_ws, size_t ws_size,
                              hipStream_t stream)
{
    (void)in_sizes; (void)n_in; (void)out_size; (void)ws_size;

    const float* act    = (const float*)d_in[0];
    const float* mask   = (const float*)d_in[1];
    const float* ln_s   = (const float*)d_in[2];
    const float* ln_o   = (const float*)d_in[3];
    const float* w_proj = (const float*)d_in[4];
    const float* w_gate = (const float*)d_in[5];
    const float* cn_s   = (const float*)d_in[6];
    const float* cn_o   = (const float*)d_in[7];
    const float* w_out  = (const float*)d_in[8];
    const float* b_out  = (const float*)d_in[9];
    const float* w_go   = (const float*)d_in[10];
    const float* b_go   = (const float*)d_in[11];
    float* out = (float*)d_out;

    char* ws = (char*)d_ws;
    size_t off = 0;
    auto alloc = [&](size_t bytes) -> char* {
        char* p = ws + off;
        off = (off + bytes + 255) & ~(size_t)255;
        return p;
    };

    _Float16* abuf    = (_Float16*)alloc((size_t)CDIM * NN * sizeof(_Float16)); // 64 MB
    _Float16* bbuf    = (_Float16*)alloc((size_t)CDIM * NN * sizeof(_Float16)); // 64 MB
    float*    gatebuf = (float*)   alloc((size_t)CDIM * NN * sizeof(float));    // 128 MB
    float*    act2buf = (float*)   alloc((size_t)CDIM * NN * sizeof(float));    // 128 MB
    _Float16* pk_proj = (_Float16*)alloc((size_t)CDIM * 2 * CDIM * sizeof(_Float16));
    _Float16* pk_gate = (_Float16*)alloc((size_t)CDIM * 2 * CDIM * sizeof(_Float16));
    _Float16* pk_out  = (_Float16*)alloc((size_t)CDIM * CDIM * sizeof(_Float16));
    _Float16* pk_go   = (_Float16*)alloc((size_t)CDIM * CDIM * sizeof(_Float16));

    // pack weights into WMMA B-fragment layout
    int totP = CDIM * 2 * CDIM;  // 32768
    int totO = CDIM * CDIM;      // 16384
    pack_w<<<(totP + 255) / 256, 256, 0, stream>>>(w_proj, pk_proj, 2 * CDIM, totP);
    pack_w<<<(totP + 255) / 256, 256, 0, stream>>>(w_gate, pk_gate, 2 * CDIM, totP);
    pack_w<<<(totO + 255) / 256, 256, 0, stream>>>(w_out,  pk_out,  CDIM,     totO);
    pack_w<<<(totO + 255) / 256, 256, 0, stream>>>(w_go,   pk_go,   CDIM,     totO);

    // fused layernorm + gated projection + output gate
    k1_ln_proj<<<NN / 16, 32, 0, stream>>>(act, mask, ln_s, ln_o, b_go,
                                           pk_proj, pk_gate, pk_go,
                                           abuf, bbuf, gatebuf);

    // batched pairwise GEMM: act2[c] = a_c @ b_c^T
    k2_pair_gemm<<<dim3(NDIM / 128, NDIM / 128, CDIM), 128, 0, stream>>>(abuf, bbuf, act2buf);

    // channel-norm + output projection + gating
    k3_cnorm_out<<<NN / 16, 32, 0, stream>>>(act2buf, gatebuf, cn_s, cn_o,
                                             b_out, pk_out, out);
}
